// SelfAtt_53403623358951
// MI455X (gfx1250) — matrix operand order repaired and checked by reading, not compiled, not run on hardware
//
#include <hip/hip_runtime.h>

// MI455X (gfx1250) flash-attention for the reference:
//   x = gather(hidden)  [8192,128] ;  Q,K,V = x @ W*  [8192,64]
//   out = softmax(Q Kt / 8) V      [8192,64]
//
// Compute-bound on the two 8192x8192x64 GEMMs (~17 GFLOP) vs ~10 MB of HBM
// traffic (0.5us at 23.3 TB/s) -> bf16 WMMA with f32 accumulation, streaming
// (flash) softmax so the 268MB score matrix never exists. K+V (2MB bf16) are
// fully L2-resident.
//
// CDNA5 features used:
//  * v_wmma_f32_16x16x32_bf16 for QK^T and P@V (S^T trick: the post-softmax
//    P tile is already in the WMMA A-operand register layout -> no transpose).
//  * tensor_load_to_lds (TDM) stages each 32x64 K tile into LDS, with the
//    descriptor's pad feature inserting +4 DWORDs per 32-DWORD row so LDS
//    rows land at 144B stride (16B-aligned, bank-conflict-free). Completion
//    via s_wait_tensorcnt; DMA overlaps the VALU-side V-transpose staging.
//  * global_prefetch (via __builtin_prefetch) on the next V tile.

typedef __attribute__((ext_vector_type(16))) __bf16 v16bf;
typedef __attribute__((ext_vector_type(8)))  __bf16 v8bf;
typedef __attribute__((ext_vector_type(8)))  float  v8f;
typedef __attribute__((ext_vector_type(4)))  float  v4f;
typedef __attribute__((ext_vector_type(4)))  unsigned int v4u;
typedef __attribute__((ext_vector_type(8)))  int    v8i;
typedef __attribute__((ext_vector_type(4)))  int    v4i;

#define NTOK 8192
#define DH   64

#if __has_builtin(__builtin_amdgcn_tensor_load_to_lds) && \
    __has_builtin(__builtin_amdgcn_s_wait_tensorcnt)
#define USE_TDM 1
#else
#define USE_TDM 0
#endif

static __device__ __forceinline__ v8f wmma_bf16(v16bf a, v16bf b, v8f c) {
  // 8 args: (neg_a, A, neg_b, B, c_mod, C, reuse_a, reuse_b)
  return __builtin_amdgcn_wmma_f32_16x16x32_bf16(false, a, false, b,
                                                 (short)0, c, false, false);
}
static __device__ __forceinline__ v16bf cat16(v8bf a, v8bf b) {
  return __builtin_shufflevector(a, b, 0,1,2,3,4,5,6,7,8,9,10,11,12,13,14,15);
}

// ---------------------------------------------------------------------------
// Phase 1: QKV projection. hidden layout is [bt][c][hw] (bt=b*16+t, hw=h*16+w),
// row n = bt*256 + hw, so x[n,c] = h[bt*32768 + c*256 + hw].
// grid = (NTOK/64, 3), block = 256: thread = (row r 0..63, d-group dg 0..3);
// lane == r -> consecutive lanes read consecutive hw: coalesced.
// Q is pre-scaled by 0.125 * log2(e) for the exp2-domain softmax.
// ---------------------------------------------------------------------------
__global__ __launch_bounds__(256) void qkv_kernel(
    const float* __restrict__ h,
    const float* __restrict__ wq, const float* __restrict__ wk,
    const float* __restrict__ wv,
    __bf16* __restrict__ q, __bf16* __restrict__ k, __bf16* __restrict__ v) {
  const float* w; __bf16* o; float scale;
  if (blockIdx.y == 0)      { w = wq; o = q; scale = 0.125f * 1.44269504f; }
  else if (blockIdx.y == 1) { w = wk; o = k; scale = 1.0f; }
  else                      { w = wv; o = v; scale = 1.0f; }

  const int r  = threadIdx.x & 63;
  const int dg = threadIdx.x >> 6;                 // 16-wide d group
  const int n  = blockIdx.x * 64 + r;
  const int bt = n >> 8, hw = n & 255;
  const float* xcol = h + bt * 32768 + hw;         // stride 256 floats over c

  float acc[16];
  #pragma unroll
  for (int i = 0; i < 16; ++i) acc[i] = 0.f;

  for (int c = 0; c < 128; ++c) {
    float xv = xcol[c * 256];
    const v4f* w4 = (const v4f*)(w + c * 64 + dg * 16);
    #pragma unroll
    for (int j = 0; j < 4; ++j) {
      v4f wj = w4[j];
      acc[4*j+0] = fmaf(xv, wj.x, acc[4*j+0]);
      acc[4*j+1] = fmaf(xv, wj.y, acc[4*j+1]);
      acc[4*j+2] = fmaf(xv, wj.z, acc[4*j+2]);
      acc[4*j+3] = fmaf(xv, wj.w, acc[4*j+3]);
    }
  }
  __bf16* orow = o + n * 64 + dg * 16;
  #pragma unroll
  for (int i = 0; i < 16; ++i) orow[i] = (__bf16)(acc[i] * scale);
}

// ---------------------------------------------------------------------------
// Phase 2: flash attention. Block = 128 threads (4 waves); wave owns 16 query
// rows; loop over 8192 keys in blocks of 32.
// Per key-block & wave: 4 WMMAs for S^T = K @ Q^T, in-register online softmax
// (one shfl_xor(16) per reduction), 4 WMMAs for P @ V.
// ---------------------------------------------------------------------------
__global__ __launch_bounds__(128) void attn_kernel(
    const __bf16* __restrict__ Q, const __bf16* __restrict__ K,
    const __bf16* __restrict__ V, float* __restrict__ out) {
  // V^T tile: 64 head-dims x 32 keys, row stride 40 bf16 (80B: 16B-aligned
  // rows, bank-conflict-free across lanes).
  __shared__ __bf16 vt[64][40];
#if USE_TDM
  // K tile staged by the TDM: 32 keys x 64 dims, TDM pad gives stride 72 bf16
  // (144B: 16B-aligned, banks 36*row mod 64 -> conflict-free).
  __shared__ __bf16 kbuf[32][72];
#endif

  const int tid  = threadIdx.x;
  const int lane = tid & 31;
  const int wvid = tid >> 5;
  const int col  = lane & 15;          // qrow (B/n index) within tile
  const int half = lane >> 4;          // lane group 0/1
  const int qbase = (blockIdx.x * 4 + wvid) * 16;

  // B operand of S^T = Q rows: lane holds qrow=col, K-slots
  // {8h..8h+7, 16+8h..16+8h+7} per 32-wide contraction chunk -> two 16B loads.
  const __bf16* qrow = Q + (qbase + col) * DH;
  v16bf B0 = cat16(*(const v8bf*)(qrow + 8*half),
                   *(const v8bf*)(qrow + 16 + 8*half));
  v16bf B1 = cat16(*(const v8bf*)(qrow + 32 + 8*half),
                   *(const v8bf*)(qrow + 48 + 8*half));

  v8f o0 = {}, o1 = {}, o2 = {}, o3 = {};      // O[16 x 64] f32 accum
  float m_cur = -3.0e38f, l_sum = 0.f;

  for (int kb = 0; kb < NTOK; kb += 32) {
    __syncthreads();                            // prev-iter LDS reads done

#if USE_TDM
    if (wvid == 0) {
      // Tensor DMA: K[kb..kb+32)[0..64) (bf16) -> kbuf, +4 DW pad / 32 DW row.
      unsigned long long ga = (unsigned long long)(uintptr_t)(K + (size_t)kb * DH);
      unsigned ldsa = (unsigned)(uintptr_t)&kbuf[0][0];   // low 32 bits = LDS offset
      v4u g0;
      g0[0] = 1u;                                          // count=1 (valid user D#)
      g0[1] = ldsa;                                        // lds_addr
      g0[2] = (unsigned)ga;                                // global_addr[31:0]
      g0[3] = (unsigned)((ga >> 32) & 0x01FFFFFFu) | 0x80000000u; // ga[56:32]|type=2
      v8i g1;
      g1[0] = (int)0x07110000u;      // mask=0, data_size=1(2B), pad_en,
                                     // pad_interval=4 (32 DW), pad_amount=3 (4 DW)
      g1[1] = (int)(64u << 16);      // tensor_dim0 = 64          (bits 63:48)
      g1[2] = (int)(8192u << 16);    // dim0 hi = 0, tensor_dim1 = 8192
      g1[3] = (int)(64u << 16);      // dim1 hi = 0, tile_dim0 = 64
      g1[4] = 32;                    // tile_dim1 = 32, tile_dim2 = 0
      g1[5] = 64;                    // tensor_dim0_stride = 64
      g1[6] = 0;                     // stride hi, tensor_dim1_stride lo
      g1[7] = 0;
      v4i g2 = {0, 0, 0, 0};         // 2-D tile: groups 2/3 unused
      v4i g3 = {0, 0, 0, 0};
      v8i gz = {0, 0, 0, 0, 0, 0, 0, 0};
      // clang-23 / therock-10.0 arity: 6 args
      // (uint32x4 g0, int32x8 g1, int32x4 g2, int32x4 g3, int32x8, i32 cpol)
      __builtin_amdgcn_tensor_load_to_lds(g0, g1, g2, g3, gz, 0);
    }
#endif

    // Stage V[32x64] transposed -> vt (all 128 threads), overlapping the TDM.
    #pragma unroll
    for (int ci = 0; ci < 2; ++ci) {
      int ch  = tid + ci * 128;                 // 0..255 chunks of 8 bf16
      int row = ch >> 3;                        // key row 0..31
      int c8  = (ch & 7) * 8;                   // head-dim base
      v8bf vz = *(const v8bf*)(V + (kb + row) * DH + c8);
      #pragma unroll
      for (int j = 0; j < 8; ++j) vt[c8 + j][row] = vz[j];
    }

    if (kb + 32 < NTOK)                         // pull next V tile into GL2
      __builtin_prefetch(V + (size_t)(kb + 32) * DH + (size_t)lane * 64, 0, 1);

#if USE_TDM
    if (wvid == 0) __builtin_amdgcn_s_wait_tensorcnt(0);
#endif
    __syncthreads();

    // A operands = K rows (A-layout: lane row = key, K-slots
    // {8h..8h+7, 16+8h..16+8h+7} per chunk -> two 16B loads each).
#if USE_TDM
    const __bf16* kr0 = &kbuf[col][0];
    const __bf16* kr1 = &kbuf[16 + col][0];
#else
    const __bf16* kr0 = K + (size_t)(kb + col) * DH;
    const __bf16* kr1 = kr0 + 16 * DH;
#endif
    v16bf A00 = cat16(*(const v8bf*)(kr0 +      8*half), *(const v8bf*)(kr0 + 16 + 8*half));
    v16bf A01 = cat16(*(const v8bf*)(kr0 + 32 + 8*half), *(const v8bf*)(kr0 + 48 + 8*half));
    v16bf A10 = cat16(*(const v8bf*)(kr1 +      8*half), *(const v8bf*)(kr1 + 16 + 8*half));
    v16bf A11 = cat16(*(const v8bf*)(kr1 + 32 + 8*half), *(const v8bf*)(kr1 + 48 + 8*half));

    v8f s0 = {}, s1 = {};                       // S^T keys 0-15 / 16-31
    s0 = wmma_bf16(A00, B0, s0);
    s0 = wmma_bf16(A01, B1, s0);
    s1 = wmma_bf16(A10, B0, s1);
    s1 = wmma_bf16(A11, B1, s1);

    // ---- online softmax over the 32 keys (stats indexed by qrow=col) ----
    float mx = -3.0e38f;
    #pragma unroll
    for (int r = 0; r < 8; ++r) mx = fmaxf(mx, fmaxf(s0[r], s1[r]));
    mx = fmaxf(mx, __shfl_xor(mx, 16, 32));
    float m_new = fmaxf(m_cur, mx);
    float alpha = exp2f(m_cur - m_new);

    float psum = 0.f;
    v16bf P = {};                               // lands in A-operand layout
    #pragma unroll
    for (int r = 0; r < 8; ++r) {
      float p0 = exp2f(s0[r] - m_new);
      float p1 = exp2f(s1[r] - m_new);
      psum += p0 + p1;
      P[r]     = (__bf16)p0;                    // key  8*half + r
      P[r + 8] = (__bf16)p1;                    // key 16 + 8*half + r
    }
    psum += __shfl_xor(psum, 16, 32);
    l_sum = l_sum * alpha + psum;
    m_cur = m_new;

    // rescale O: row m = half*8 + r needs alpha held at lane m
    #pragma unroll
    for (int r = 0; r < 8; ++r) {
      float ar = __shfl(alpha, half * 8 + r, 32);
      o0[r] *= ar; o1[r] *= ar; o2[r] *= ar; o3[r] *= ar;
    }

    // P @ V: B operand from vt (lane holds head-dim n, keys 8*half..+7, 16+8*half..+7)
    const __bf16* vr0 = &vt[ 0 + col][8*half];
    const __bf16* vr1 = &vt[16 + col][8*half];
    const __bf16* vr2 = &vt[32 + col][8*half];
    const __bf16* vr3 = &vt[48 + col][8*half];
    o0 = wmma_bf16(P, cat16(*(const v8bf*)vr0, *(const v8bf*)(vr0 + 16)), o0);
    o1 = wmma_bf16(P, cat16(*(const v8bf*)vr1, *(const v8bf*)(vr1 + 16)), o1);
    o2 = wmma_bf16(P, cat16(*(const v8bf*)vr2, *(const v8bf*)(vr2 + 16)), o2);
    o3 = wmma_bf16(P, cat16(*(const v8bf*)vr3, *(const v8bf*)(vr3 + 16)), o3);
  }

  // normalize and write: lane holds column col of rows m = half*8 + r
  #pragma unroll
  for (int r = 0; r < 8; ++r) {
    int m = half * 8 + r;
    float lr  = __shfl(l_sum, m, 32);
    float inv = 1.0f / lr;
    float* orow = out + (size_t)(qbase + m) * DH + col;
    orow[0]  = o0[r] * inv;
    orow[16] = o1[r] * inv;
    orow[32] = o2[r] * inv;
    orow[48] = o3[r] * inv;
  }
}

// ---------------------------------------------------------------------------
extern "C" void kernel_launch(void* const* d_in, const int* in_sizes, int n_in,
                              void* d_out, int out_size, void* d_ws, size_t ws_size,
                              hipStream_t stream) {
  const float* h  = (const float*)d_in[0];
  const float* wq = (const float*)d_in[1];
  const float* wk = (const float*)d_in[2];
  const float* wv = (const float*)d_in[3];

  // workspace: Q,K,V as bf16 [8192 x 64] each (3 MB total)
  __bf16* q = (__bf16*)d_ws;
  __bf16* k = q + (size_t)NTOK * DH;
  __bf16* v = k + (size_t)NTOK * DH;

  qkv_kernel<<<dim3(NTOK / 64, 3), 256, 0, stream>>>(h, wq, wk, wv, q, k, v);
  attn_kernel<<<NTOK / 64, 128, 0, stream>>>(q, k, v, (float*)d_out);
}
